// MultiHeadAttention_33122787787233
// MI455X (gfx1250) — matrix operand order repaired and checked by reading, not compiled, not run on hardware
//
#include <hip/hip_runtime.h>
#include <math.h>

// ---------------------------------------------------------------------------
// MHA on gfx1250 (MI455X):  all GEMMs via v_wmma_f32_16x16x32_f16 (wave32).
//   D_MODEL=1024, HEADS=16, D_K=64, B=2, S=2048  ->  M = B*S = 4096 rows.
// GEMM: 128x64 WG tile, 32x64 per-wave tile; weight tile double-buffered in
// LDS via global_load_async_to_lds_b128 (ASYNCcnt) overlapping the WMMAs.
// ---------------------------------------------------------------------------

#define D_MODEL 1024
#define HEADS   16
#define D_K     64
#define BATCH   2
#define SEQ     2048
#define MROWS   (BATCH * SEQ)     // 4096
#define NEGVAL  (-1e9f)

typedef _Float16 v4h  __attribute__((ext_vector_type(4)));
typedef _Float16 v8h  __attribute__((ext_vector_type(8)));
typedef _Float16 v16h __attribute__((ext_vector_type(16)));
typedef float    v4f  __attribute__((ext_vector_type(4)));
typedef float    v8f  __attribute__((ext_vector_type(8)));
typedef int      v4i  __attribute__((ext_vector_type(4)));

static __device__ __forceinline__ v16h cat16(v8h lo, v8h hi) {
  return __builtin_shufflevector(lo, hi, 0,1,2,3,4,5,6,7,8,9,10,11,12,13,14,15);
}

static __device__ __forceinline__ v8f wmma_f16(v16h a, v16h b, v8f c) {
  // D = A(16x32 f16) x B(32x16 f16) + C(16x16 f32)
  return __builtin_amdgcn_wmma_f32_16x16x32_f16(false, a, false, b,
                                                (short)0, c, false, false);
}

// Async copy of 16 bytes Global -> LDS (tracked by ASYNCcnt, not LOADcnt).
// Per ISA 15.18.3: LDS address from VDST, global address from VADDR, same offset added to both.
static __device__ __forceinline__ void async_b128(unsigned lds_addr,
                                                  unsigned long long gaddr) {
  asm volatile("global_load_async_to_lds_b128 %0, %1, off"
               :: "v"(lds_addr), "v"(gaddr) : "memory");
}

// ---------------------------------------------------------------------------
// fp32 -> f16 convert (4-wide)
// ---------------------------------------------------------------------------
__global__ __launch_bounds__(256) void cvt_f32_to_f16(
    const float* __restrict__ src, _Float16* __restrict__ dst, int n) {
  int stride = gridDim.x * blockDim.x * 4;
  for (int i = (blockIdx.x * blockDim.x + threadIdx.x) * 4; i < n; i += stride) {
    v4f f = *(const v4f*)(src + i);
    v4h h;
    h[0] = (_Float16)f[0]; h[1] = (_Float16)f[1];
    h[2] = (_Float16)f[2]; h[3] = (_Float16)f[3];
    *(v4h*)(dst + i) = h;
  }
}

// ---------------------------------------------------------------------------
// GEMM: Y = X(MxK) * W(NxK)^T   (torch Linear convention)
// Block = 128 threads (4 waves). WG tile 128(m) x 64(n); wave tile 32 x 64.
// The 64x32 weight tile (shared by all 4 waves) is async-staged into LDS,
// double-buffered; A fragments stream from global (per-wave-private rows).
// Steady-state loop has no branches: last k-step is peeled.
// mode 1: f16 row-major out (ld=N), scaled   (used for Q (scale=1/8) and K)
// mode 2: f16 "V transposed head-major": Vt[(b*1024 + n)*S + s]
// mode 3: fp32 row-major out (final output)
// ---------------------------------------------------------------------------
__global__ __launch_bounds__(128) void gemm_xwT(
    const _Float16* __restrict__ X, const _Float16* __restrict__ W,
    void* __restrict__ out, int Ndim, int Kdim, int mode, float scale) {
  __shared__ _Float16 wsm[2][64][32];     // 2 x 4KB double buffer

  const int wave = threadIdx.x >> 5;
  const int lane = threadIdx.x & 31;
  const int r    = lane & 15;
  const int hi   = lane >> 4;
  const int m0   = blockIdx.y * 128 + wave * 32;
  const int n0   = blockIdx.x * 64;

  // staging map: thread t copies 16B: row = t&63, halves (t>>6)*16 .. (t>>6)*16+7
  const int srow = threadIdx.x & 63;
  const int sseg = threadIdx.x >> 6;
  const _Float16* gWrow = W + (size_t)(n0 + srow) * Kdim + sseg * 16;
  const unsigned ldsA[2] = {
      (unsigned)(uintptr_t)&wsm[0][srow][sseg * 16],
      (unsigned)(uintptr_t)&wsm[1][srow][sseg * 16]};

  const v8f z8 = {0.f,0.f,0.f,0.f,0.f,0.f,0.f,0.f};
  v8f acc0[4] = {z8, z8, z8, z8};
  v8f acc1[4] = {z8, z8, z8, z8};

  const _Float16* Arow0 = X + (size_t)(m0 + r) * Kdim;
  const _Float16* Arow1 = X + (size_t)(m0 + 16 + r) * Kdim;

  // one k-step of compute against LDS buffer `buf`
  auto kstep = [&](int buf, v16h a0, v16h a1) {
    v16h bfr[4];
#pragma unroll
    for (int nc = 0; nc < 4; ++nc) {          // issue all 8 ds_load_b128 first
      const _Float16* brow = &wsm[buf][nc * 16 + r][hi * 8];
      bfr[nc] = cat16(*(const v8h*)brow, *(const v8h*)(brow + 16));
    }
#pragma unroll
    for (int nc = 0; nc < 4; ++nc) {          // then 8 back-to-back WMMAs
      acc0[nc] = wmma_f16(a0, bfr[nc], acc0[nc]);
      acc1[nc] = wmma_f16(a1, bfr[nc], acc1[nc]);
    }
  };

  // prologue: stage k0 = 0 into buffer 0
  async_b128(ldsA[0], (unsigned long long)(uintptr_t)gWrow);

  int buf = 0;
  for (int k0 = 0; k0 < Kdim - 32; k0 += 32) {
    // A fragments for the current step: in flight across the barrier wait
    v16h a0 = cat16(*(const v8h*)(Arow0 + k0 + hi * 8),
                    *(const v8h*)(Arow0 + k0 + hi * 8 + 16));
    v16h a1 = cat16(*(const v8h*)(Arow1 + k0 + hi * 8),
                    *(const v8h*)(Arow1 + k0 + hi * 8 + 16));
    __builtin_prefetch(Arow0 + k0 + 128, 0, 3);

    // stage next tile into the other buffer, then wait for the current tile
    async_b128(ldsA[buf ^ 1],
               (unsigned long long)(uintptr_t)(gWrow + k0 + 32));
    asm volatile("s_wait_asynccnt 0x1" ::: "memory");
    __syncthreads();                           // all waves' async data visible

    kstep(buf, a0, a1);

    __syncthreads();                           // reads done before overwrite
    buf ^= 1;
  }

  // peeled last k-step (no further staging, no trailing barrier)
  {
    const int k0 = Kdim - 32;
    v16h a0 = cat16(*(const v8h*)(Arow0 + k0 + hi * 8),
                    *(const v8h*)(Arow0 + k0 + hi * 8 + 16));
    v16h a1 = cat16(*(const v8h*)(Arow1 + k0 + hi * 8),
                    *(const v8h*)(Arow1 + k0 + hi * 8 + 16));
    asm volatile("s_wait_asynccnt 0x0" ::: "memory");
    __syncthreads();
    kstep(buf, a0, a1);
  }

  // C layout: lane col n = r; vgpr j -> row m = m0 + t*16 + hi*8 + j
  if (mode == 1) {
    _Float16* O = (_Float16*)out;
#pragma unroll
    for (int t = 0; t < 2; ++t) {
      const v8f* acc = t ? acc1 : acc0;
#pragma unroll
      for (int nc = 0; nc < 4; ++nc) {
        int n = n0 + nc * 16 + r;
#pragma unroll
        for (int j = 0; j < 8; ++j) {
          int m = m0 + t * 16 + hi * 8 + j;
          O[(size_t)m * Ndim + n] = (_Float16)(acc[nc][j] * scale);
        }
      }
    }
  } else if (mode == 2) {
    _Float16* O = (_Float16*)out;
#pragma unroll
    for (int t = 0; t < 2; ++t) {
      const v8f* acc = t ? acc1 : acc0;
      const int mbase = m0 + t * 16 + hi * 8;
      const int b = mbase / SEQ;
      const int s = mbase % SEQ;
#pragma unroll
      for (int nc = 0; nc < 4; ++nc) {
        int n = n0 + nc * 16 + r;
        v8h v;
#pragma unroll
        for (int j = 0; j < 8; ++j) v[j] = (_Float16)(acc[nc][j] * scale);
        *(v8h*)(O + (size_t)(b * 1024 + n) * SEQ + s) = v;   // 16B aligned
      }
    }
  } else {
    float* O = (float*)out;
#pragma unroll
    for (int t = 0; t < 2; ++t) {
      const v8f* acc = t ? acc1 : acc0;
#pragma unroll
      for (int nc = 0; nc < 4; ++nc) {
        int n = n0 + nc * 16 + r;
#pragma unroll
        for (int j = 0; j < 8; ++j) {
          int m = m0 + t * 16 + hi * 8 + j;
          O[(size_t)m * Ndim + n] = acc[nc][j] * scale;
        }
      }
    }
  }
}

// ---------------------------------------------------------------------------
// Register-resident flash attention (transposed formulation).
//   S^T(keys x q) = K_blk(16x64) x Q^T(64x16)   -> one query per lane
//   O^T(d x q)   += Vt(64x32) x P^T(32x16)
// Grid: (S/64, B*H); 4 waves, each owns 16 queries for the whole key loop.
// Qh/Kh: f16 row-major [B*S][1024] (head at col h*64).  Vt: [(b*1024+h*64+d)][S].
// ---------------------------------------------------------------------------
__global__ __launch_bounds__(128) void flash_attn_wmma(
    const _Float16* __restrict__ Qh, const _Float16* __restrict__ Kh,
    const _Float16* __restrict__ Vt, const int* __restrict__ mask,
    _Float16* __restrict__ Xout) {
  const int wave = threadIdx.x >> 5;
  const int lane = threadIdx.x & 31;
  const int r    = lane & 15;
  const int hi   = lane >> 4;
  const bool lo16 = (hi == 0);
  const int bh = blockIdx.y;
  const int b  = bh >> 4;
  const int h  = bh & 15;
  const int q0 = blockIdx.x * 64 + wave * 16;
  const int q  = q0 + r;                    // this lane's query (fixed)

  // Q^T as B operand: B^T row q = Qh row q (contiguous over d). 1/8 scale is
  // pre-folded into Qh by the Q-projection epilogue.
  const _Float16* Qrow = Qh + (size_t)(b * SEQ + q) * D_MODEL + h * D_K;
  const v16h bq0 = cat16(*(const v8h*)(Qrow + hi * 8),
                         *(const v8h*)(Qrow + hi * 8 + 16));
  const v16h bq1 = cat16(*(const v8h*)(Qrow + 32 + hi * 8),
                         *(const v8h*)(Qrow + 32 + hi * 8 + 16));

  const v8f z8 = {0.f,0.f,0.f,0.f,0.f,0.f,0.f,0.f};
  v8f o[4] = {z8, z8, z8, z8};              // O^T: 64 d-rows x 16 q-cols
  float m_run = -1e30f, l_run = 0.f;
  const int* mrow = mask + (size_t)q * SEQ;

  for (int kb = 0; kb < SEQ; kb += 32) {
    // ---- scores: two 16-key tiles as WMMA A operands (K rows contiguous) ----
    const _Float16* K0 = Kh + (size_t)(b * SEQ + kb + r) * D_MODEL + h * D_K + hi * 8;
    const _Float16* K1 = K0 + (size_t)16 * D_MODEL;
    __builtin_prefetch(K0 + (size_t)32 * D_MODEL, 0, 3);   // next key block
    v8f st0 = z8, st1 = z8;
    st0 = wmma_f16(cat16(*(const v8h*)K0,        *(const v8h*)(K0 + 16)), bq0, st0);
    st0 = wmma_f16(cat16(*(const v8h*)(K0 + 32), *(const v8h*)(K0 + 48)), bq1, st0);
    st1 = wmma_f16(cat16(*(const v8h*)K1,        *(const v8h*)(K1 + 16)), bq0, st1);
    st1 = wmma_f16(cat16(*(const v8h*)(K1 + 32), *(const v8h*)(K1 + 48)), bq1, st1);

    // ---- mask (vgpr j of st0 -> key kb + hi*8 + j; st1 -> +16) ----
    int mk0[8], mk1[8];
    *(v4i*)&mk0[0] = *(const v4i*)(mrow + kb + hi * 8);
    *(v4i*)&mk0[4] = *(const v4i*)(mrow + kb + hi * 8 + 4);
    *(v4i*)&mk1[0] = *(const v4i*)(mrow + kb + 16 + hi * 8);
    *(v4i*)&mk1[4] = *(const v4i*)(mrow + kb + 16 + hi * 8 + 4);
    float s0[8], s1[8];
#pragma unroll
    for (int j = 0; j < 8; ++j) {
      s0[j] = (mk0[j] == 0) ? NEGVAL : st0[j];
      s1[j] = (mk1[j] == 0) ? NEGVAL : st1[j];
    }

    // ---- online softmax: lane holds 16 of 32 keys, partner (xor16) the rest
    float bm = -1e30f;
#pragma unroll
    for (int j = 0; j < 8; ++j) bm = fmaxf(bm, fmaxf(s0[j], s1[j]));
    bm = fmaxf(bm, __shfl_xor(bm, 16, 32));
    const float m_new = fmaxf(m_run, bm);
    const float alpha = __expf(m_run - m_new);
    float p0[8], p1[8], lsum = 0.f;
#pragma unroll
    for (int j = 0; j < 8; ++j) {
      p0[j] = __expf(s0[j] - m_new);
      p1[j] = __expf(s1[j] - m_new);
      lsum += p0[j] + p1[j];
    }
    lsum += __shfl_xor(lsum, 16, 32);
    l_run = l_run * alpha + lsum;
    m_run = m_new;

    // ---- P^T as B operand: lane<16 needs keys kb+0..15 (own + partner of
    //      tile0); lane>=16 needs kb+16..31 (partner + own of tile1).
    v16h bp;
#pragma unroll
    for (int j = 0; j < 8; ++j) {
      float t0o = __shfl_xor(p0[j], 16, 32);
      float t1o = __shfl_xor(p1[j], 16, 32);
      bp[j]     = (_Float16)(lo16 ? p0[j] : t1o);
      bp[8 + j] = (_Float16)(lo16 ? t0o   : p1[j]);
    }

    // ---- rescale O^T, accumulate O^T += Vt x P^T (Vt rows contiguous) ----
#pragma unroll
    for (int dc = 0; dc < 4; ++dc)
#pragma unroll
      for (int j = 0; j < 8; ++j) o[dc][j] *= alpha;
#pragma unroll
    for (int dc = 0; dc < 4; ++dc) {
      const _Float16* Vr =
          Vt + (size_t)(bh * 64 + dc * 16 + r) * SEQ + kb + hi * 16;
      v16h av = cat16(*(const v8h*)Vr, *(const v8h*)(Vr + 8));
      o[dc] = wmma_f16(av, bp, o[dc]);
    }
  }

  // ---- normalize and store f16 row-major (8 consecutive d per lane) ----
  const float inv = 1.f / l_run;
  _Float16* Orow = Xout + (size_t)(b * SEQ + q) * D_MODEL + h * D_K;
#pragma unroll
  for (int dc = 0; dc < 4; ++dc) {
    v8h v;
#pragma unroll
    for (int j = 0; j < 8; ++j) v[j] = (_Float16)(o[dc][j] * inv);
    *(v8h*)(Orow + dc * 16 + hi * 8) = v;
  }
}

// ---------------------------------------------------------------------------
extern "C" void kernel_launch(void* const* d_in, const int* in_sizes, int n_in,
                              void* d_out, int out_size, void* d_ws,
                              size_t ws_size, hipStream_t stream) {
  const float* Query = (const float*)d_in[0];
  const float* Key   = (const float*)d_in[1];
  const float* Value = (const float*)d_in[2];
  const int*   mask  = (const int*)d_in[3];
  const float* W_q   = (const float*)d_in[4];
  const float* W_k   = (const float*)d_in[5];
  const float* W_v   = (const float*)d_in[6];
  const float* W_o   = (const float*)d_in[7];

  char* ws = (char*)d_ws;
  size_t off = 0;
  auto take = [&](size_t bytes) -> void* {
    void* p = ws + off;
    off += (bytes + 255) & ~(size_t)255;
    return p;
  };

  const size_t MD = (size_t)MROWS * D_MODEL;   // 4,194,304 elements
  const size_t DD = (size_t)D_MODEL * D_MODEL; // 1,048,576 elements

  _Float16* Xq  = (_Float16*)take(MD * 2);
  _Float16* Xk  = (_Float16*)take(MD * 2);
  _Float16* Xv  = (_Float16*)take(MD * 2);
  _Float16* Wqh = (_Float16*)take(DD * 2);
  _Float16* Wkh = (_Float16*)take(DD * 2);
  _Float16* Wvh = (_Float16*)take(DD * 2);
  _Float16* Woh = (_Float16*)take(DD * 2);
  _Float16* Qhb = (_Float16*)take(MD * 2);
  _Float16* Khb = (_Float16*)take(MD * 2);
  _Float16* Vtb = (_Float16*)take(MD * 2);
  _Float16* Xat = Xq;  // reuse: Xq is dead after the Q projection

  // 1) fp32 -> f16 conversions
  cvt_f32_to_f16<<<2048, 256, 0, stream>>>(Query, Xq, (int)MD);
  cvt_f32_to_f16<<<2048, 256, 0, stream>>>(Key,   Xk, (int)MD);
  cvt_f32_to_f16<<<2048, 256, 0, stream>>>(Value, Xv, (int)MD);
  cvt_f32_to_f16<<<1024, 256, 0, stream>>>(W_q, Wqh, (int)DD);
  cvt_f32_to_f16<<<1024, 256, 0, stream>>>(W_k, Wkh, (int)DD);
  cvt_f32_to_f16<<<1024, 256, 0, stream>>>(W_v, Wvh, (int)DD);
  cvt_f32_to_f16<<<1024, 256, 0, stream>>>(W_o, Woh, (int)DD);

  // 2) projections (WMMA).  Q gets 1/sqrt(d_k) = 0.125 folded in.
  dim3 ggrid(D_MODEL / 64, MROWS / 128);  // (16, 32)
  gemm_xwT<<<ggrid, 128, 0, stream>>>(Xq, Wqh, Qhb, D_MODEL, D_MODEL, 1, 0.125f);
  gemm_xwT<<<ggrid, 128, 0, stream>>>(Xk, Wkh, Khb, D_MODEL, D_MODEL, 1, 1.0f);
  gemm_xwT<<<ggrid, 128, 0, stream>>>(Xv, Wvh, Vtb, D_MODEL, D_MODEL, 2, 1.0f);

  // 3) attention (WMMA flash, registers only)
  flash_attn_wmma<<<dim3(SEQ / 64, BATCH * HEADS), 128, 0, stream>>>(
      Qhb, Khb, Vtb, mask, Xat);

  // 4) output projection -> fp32 d_out
  gemm_xwT<<<ggrid, 128, 0, stream>>>(Xat, Woh, d_out, D_MODEL, D_MODEL, 3, 1.0f);
}
